// IntelligentRouter_44659069944354
// MI455X (gfx1250) — matrix operand order repaired and checked by reading, not compiled, not run on hardware
//
#include <hip/hip_runtime.h>

// ---------- types ----------
typedef unsigned short u16;
typedef __attribute__((ext_vector_type(16))) __bf16        v16bf;
typedef __attribute__((ext_vector_type(8)))  float         v8f;
typedef __attribute__((ext_vector_type(4)))  unsigned int  u32x4;

union Frag { v16bf v; u32x4 q[2]; };
union Acc  { v8f  v; float f[8]; };

// ---------- constants ----------
#define NB   16384
#define ND   2048
#define NE   16
#define NH   1024
#define NDE  2064
#define TOPK 409.0f

// d_ws byte offsets
#define OFF_XB    ((size_t)0)          // x bf16          [16384,2048]  64 MiB
#define OFF_WS1T  ((size_t)67108864)   // Ws1^T bf16      [1024,2048]    4 MiB
#define OFF_WR1T  ((size_t)71303168)   // Wr1^T bf16      [1024,2064]  ~4 MiB
#define OFF_WS2T  ((size_t)75530240)   // Ws2^T bf16      [16,1024]
#define OFF_WR2T  ((size_t)75563008)   // Wr2^T bf16      [16,1024]
#define OFF_PATB  ((size_t)75595776)   // patterns bf16   [16,2048] (already N-major)
#define OFF_SCR   ((size_t)75661312)   // scores bf16     [16384,16]
#define OFF_H     ((size_t)76185600)   // h bf16          [16384,1024]  32 MiB (h1 then h2)
#define OFF_KVAL  ((size_t)109740032)  // int k

// d_out float offsets: gates[16384,4], idx[16384,4], probs[16384,16], stats[16384,6]
#define OOF_GATES 0
#define OOF_IDX   65536
#define OOF_PROBS 131072
#define OOF_STATS 393216

// ---------- helpers ----------
__device__ __forceinline__ u16 f2bf(float f) {
  union { float f; unsigned int u; } c; c.f = f;
  unsigned int u = c.u;
  u += 0x7fffu + ((u >> 16) & 1u);   // round-to-nearest-even
  return (u16)(u >> 16);
}

__device__ __forceinline__ v8f wmma_bf16(const Frag& a, const Frag& b, v8f c) {
  return __builtin_amdgcn_wmma_f32_16x16x32_bf16(false, a.v, false, b.v,
                                                 (short)0, c, false, false);
}

__device__ __forceinline__ float block_reduce(float v, float* red, bool ismax) {
  int tid = threadIdx.x;
  red[tid] = v; __syncthreads();
  for (int s = 128; s > 0; s >>= 1) {
    if (tid < s) red[tid] = ismax ? fmaxf(red[tid], red[tid + s]) : (red[tid] + red[tid + s]);
    __syncthreads();
  }
  float r = red[0]; __syncthreads();
  return r;
}

__device__ __forceinline__ void store_tile_relu_bf16(v8f c, int mb, int nb, float bv,
                                                     u16* out, int ldo, int hi, int l15) {
  Acc u; u.v = c;
#pragma unroll
  for (int r = 0; r < 8; ++r) {
    int m = mb + r + hi * 8;
    float val = fmaxf(u.f[r] + bv, 0.f);
    out[(size_t)m * ldo + nb + l15] = f2bf(val);
  }
}

__device__ __forceinline__ void mma_loop(const u16* ap, const u16* bp, int K, v8f& c) {
  for (int k0 = 0; k0 < K; k0 += 32) {
    Frag a, b;
    a.q[0] = *(const u32x4*)(ap + k0);
    a.q[1] = *(const u32x4*)(ap + k0 + 16);
    b.q[0] = *(const u32x4*)(bp + k0);
    b.q[1] = *(const u32x4*)(bp + k0 + 16);
    c = wmma_bf16(a, b, c);
  }
}

// ---------- conversion kernels ----------
__global__ __launch_bounds__(256) void cvt_kernel(const float* __restrict__ in,
                                                  u16* __restrict__ out, int n) {
  int i = blockIdx.x * 256 + threadIdx.x;
  int stride = gridDim.x * 256;
  for (; i < n; i += stride) out[i] = f2bf(in[i]);
}

// in[K,N] row-major -> out[N,K] bf16
__global__ __launch_bounds__(256) void cvtT_kernel(const float* __restrict__ in,
                                                   u16* __restrict__ out, int K, int N) {
  int i = blockIdx.x * 256 + threadIdx.x;
  int stride = gridDim.x * 256;
  int total = K * N;
  for (; i < total; i += stride) {
    int n = i / K, k = i - n * K;
    out[i] = f2bf(in[(size_t)k * N + n]);
  }
}

// ---------- per-row feature statistics ----------
__global__ __launch_bounds__(256) void stats_kernel(const float* __restrict__ x,
                                                    float* __restrict__ outStats) {
  __shared__ float row[2048];
  __shared__ float red[256];
  const int tid = threadIdx.x;
  const float* xr = x + (size_t)blockIdx.x * ND;

  float lsum = 0, lsq = 0, labs = 0, lmax = 0, lzero = 0;
#pragma unroll
  for (int i = 0; i < 8; ++i) {
    float v = xr[tid + i * 256];
    row[tid + i * 256] = v;
    lsum += v; lsq += v * v;
    float a = fabsf(v);
    labs += a; lmax = fmaxf(lmax, a);
    lzero += (v == 0.f) ? 1.f : 0.f;
  }
  __syncthreads();
  float sum    = block_reduce(lsum, red, false);
  float sumsq  = block_reduce(lsq,  red, false);
  float sumabs = block_reduce(labs, red, false);
  float maxabs = block_reduce(lmax, red, true);
  float zeros  = block_reduce(lzero, red, false);

  float mean  = sum / (float)ND;
  float var_u = (sumsq - (float)ND * mean * mean) / (float)(ND - 1);
  float stdv  = sqrtf(var_u + 1e-8f);

  float lsk = 0.f;
#pragma unroll
  for (int i = 0; i < 8; ++i) {
    float z = (row[tid + i * 256] - mean) / stdv;
    lsk += z * z * z;
  }
  float skew = block_reduce(lsk, red, false) / (float)ND;

  // top-409 |x| sum via threshold binary search (t uniform across block)
  float lo = 0.f, hi = maxabs;
  for (int it = 0; it < 32; ++it) {
    float mid = 0.5f * (lo + hi);
    float lc = 0.f;
#pragma unroll
    for (int i = 0; i < 8; ++i) lc += (fabsf(row[tid + i * 256]) > mid) ? 1.f : 0.f;
    float c = block_reduce(lc, red, false);
    if (c > TOPK) lo = mid; else hi = mid;
  }
  float ls = 0.f, lc2 = 0.f;
#pragma unroll
  for (int i = 0; i < 8; ++i) {
    float a = fabsf(row[tid + i * 256]);
    if (a > hi) { ls += a; lc2 += 1.f; }
  }
  float S = block_reduce(ls, red, false);
  float C = block_reduce(lc2, red, false);
  float top = S + fmaxf(0.f, TOPK - C) * hi;

  if (tid == 0) {
    float* o = outStats + (size_t)blockIdx.x * 6;
    o[0] = zeros / (float)ND;          // sparsity
    o[1] = var_u;                      // variance (unbiased)
    o[2] = maxabs;                     // magnitude
    o[3] = sqrtf(sumsq);               // norm
    o[4] = skew;                       // skewness
    o[5] = top / (sumabs + 1e-8f);     // concentration
  }
}

// ---------- k predictor + lower median (single workgroup) ----------
__global__ __launch_bounds__(256) void kpred_kernel(const float* __restrict__ stats,
                                                    const float* __restrict__ Wk1,
                                                    const float* __restrict__ bk1,
                                                    const float* __restrict__ Wk2,
                                                    const float* __restrict__ bk2,
                                                    int* __restrict__ kout) {
  __shared__ float red[256];
  __shared__ float w1[96], b1[16], w2[16];
  int tid = threadIdx.x;
  if (tid < 96) w1[tid] = Wk1[tid];
  if (tid < 16) { b1[tid] = bk1[tid]; w2[tid] = Wk2[tid]; }
  __syncthreads();
  float b2 = bk2[0];
  float c1 = 0, c2 = 0, c3 = 0;
  for (int r = tid; r < NB; r += 256) {
    const float* s = stats + (size_t)r * 6;
    float acc = b2;
#pragma unroll
    for (int j = 0; j < 16; ++j) {
      float h = b1[j];
#pragma unroll
      for (int i = 0; i < 6; ++i) h += s[i] * w1[i * 16 + j];
      acc += fmaxf(h, 0.f) * w2[j];
    }
    float kv = 1.f + 3.f / (1.f + __expf(-acc));
    c1 += (kv < 2.f) ? 1.f : 0.f;
    c2 += (kv < 3.f) ? 1.f : 0.f;
    c3 += (kv < 4.f) ? 1.f : 0.f;
  }
  float C1 = block_reduce(c1, red, false);
  float C2 = block_reduce(c2, red, false);
  float C3 = block_reduce(c3, red, false);
  if (tid == 0) {
    const float m = (float)((NB - 1) / 2);  // 8191: lower-median rank
    int k = (m < C1) ? 1 : (m < C2) ? 2 : (m < C3) ? 3 : 4;
    *kout = k;
  }
}

// ---------- big GEMM: out = relu(A @ B + bias) in bf16, optional K=16 tail ----------
// A [M,K] bf16 row-major, BT [N,K] bf16 row-major (B transposed).
// Block tile 128(M) x 128(N): 8 waves as 4(M) x 2(N), each wave 32x64
// (2 M-subtiles x 4 N-subtiles = 8 WMMAs per 32-K step, 12 b128 loads).
__global__ __launch_bounds__(256) void gemm_big_kernel(const u16* __restrict__ A, int lda,
                                                       const u16* __restrict__ BT, int ldb,
                                                       const float* __restrict__ bias,
                                                       u16* __restrict__ out, int ldo, int K,
                                                       const u16* __restrict__ tailA,
                                                       const u16* __restrict__ tailBT) {
  const int lane = threadIdx.x & 31;
  const int wid  = threadIdx.x >> 5;
  const int wm = wid & 3, wn = wid >> 2;
  const int l15 = lane & 15, hi = lane >> 4;
  const int m0 = blockIdx.x * 128 + wm * 32;
  const int n0 = blockIdx.y * 128 + wn * 64;

  const u16* a0p = A  + (size_t)(m0 + l15) * lda + hi * 16;
  const u16* a1p = a0p + (size_t)16 * lda;
  const u16* b0p = BT + (size_t)(n0 + l15) * ldb + hi * 16;
  const u16* b1p = b0p + (size_t)16 * ldb;
  const u16* b2p = b0p + (size_t)32 * ldb;
  const u16* b3p = b0p + (size_t)48 * ldb;

  v8f c00 = {}, c01 = {}, c02 = {}, c03 = {};
  v8f c10 = {}, c11 = {}, c12 = {}, c13 = {};

  for (int k0 = 0; k0 < K; k0 += 32) {
    Frag a0, a1, b0, b1, b2, b3;
    const u16* pa0 = a0p + k0;
    const u16* pa1 = a1p + k0;
    a0.q[0] = *(const u32x4*)(pa0);      a0.q[1] = *(const u32x4*)(pa0 + 8);
    a1.q[0] = *(const u32x4*)(pa1);      a1.q[1] = *(const u32x4*)(pa1 + 8);
    const u16* pb0 = b0p + k0;
    const u16* pb1 = b1p + k0;
    const u16* pb2 = b2p + k0;
    const u16* pb3 = b3p + k0;
    b0.q[0] = *(const u32x4*)(pb0);      b0.q[1] = *(const u32x4*)(pb0 + 8);
    b1.q[0] = *(const u32x4*)(pb1);      b1.q[1] = *(const u32x4*)(pb1 + 8);
    b2.q[0] = *(const u32x4*)(pb2);      b2.q[1] = *(const u32x4*)(pb2 + 8);
    b3.q[0] = *(const u32x4*)(pb3);      b3.q[1] = *(const u32x4*)(pb3 + 8);
    __builtin_prefetch((const void*)(pa0 + 512), 0, 1);
    c00 = wmma_bf16(a0, b0, c00);
    c01 = wmma_bf16(a0, b1, c01);
    c02 = wmma_bf16(a0, b2, c02);
    c03 = wmma_bf16(a0, b3, c03);
    c10 = wmma_bf16(a1, b0, c10);
    c11 = wmma_bf16(a1, b1, c11);
    c12 = wmma_bf16(a1, b2, c12);
    c13 = wmma_bf16(a1, b3, c13);
  }

  if (tailA) {  // K=16 tail (scores @ Wr1[D:,:]) zero-padded to K=32
    u32x4 z = {0u, 0u, 0u, 0u};
    Frag a0, a1, b0, b1, b2, b3;
    const u16* ta0 = tailA + (size_t)(m0 + l15) * 16;
    const u16* ta1 = ta0 + (size_t)16 * 16;
    a0.q[0] = hi ? z : *(const u32x4*)(ta0); a0.q[1] = hi ? z : *(const u32x4*)(ta0 + 8);
    a1.q[0] = hi ? z : *(const u32x4*)(ta1); a1.q[1] = hi ? z : *(const u32x4*)(ta1 + 8);
    if (hi == 0) {
      const u16* tb0 = tailBT + (size_t)(n0 + l15) * ldb;
      const u16* tb1 = tb0 + (size_t)16 * ldb;
      const u16* tb2 = tb0 + (size_t)32 * ldb;
      const u16* tb3 = tb0 + (size_t)48 * ldb;
      b0.q[0] = *(const u32x4*)(tb0); b0.q[1] = *(const u32x4*)(tb0 + 8);
      b1.q[0] = *(const u32x4*)(tb1); b1.q[1] = *(const u32x4*)(tb1 + 8);
      b2.q[0] = *(const u32x4*)(tb2); b2.q[1] = *(const u32x4*)(tb2 + 8);
      b3.q[0] = *(const u32x4*)(tb3); b3.q[1] = *(const u32x4*)(tb3 + 8);
    } else {
      b0.q[0] = z; b0.q[1] = z; b1.q[0] = z; b1.q[1] = z;
      b2.q[0] = z; b2.q[1] = z; b3.q[0] = z; b3.q[1] = z;
    }
    c00 = wmma_bf16(a0, b0, c00);
    c01 = wmma_bf16(a0, b1, c01);
    c02 = wmma_bf16(a0, b2, c02);
    c03 = wmma_bf16(a0, b3, c03);
    c10 = wmma_bf16(a1, b0, c10);
    c11 = wmma_bf16(a1, b1, c11);
    c12 = wmma_bf16(a1, b2, c12);
    c13 = wmma_bf16(a1, b3, c13);
  }

  float bn0 = bias[n0 + l15];
  float bn1 = bias[n0 + 16 + l15];
  float bn2 = bias[n0 + 32 + l15];
  float bn3 = bias[n0 + 48 + l15];
  store_tile_relu_bf16(c00, m0,      n0,      bn0, out, ldo, hi, l15);
  store_tile_relu_bf16(c01, m0,      n0 + 16, bn1, out, ldo, hi, l15);
  store_tile_relu_bf16(c02, m0,      n0 + 32, bn2, out, ldo, hi, l15);
  store_tile_relu_bf16(c03, m0,      n0 + 48, bn3, out, ldo, hi, l15);
  store_tile_relu_bf16(c10, m0 + 16, n0,      bn0, out, ldo, hi, l15);
  store_tile_relu_bf16(c11, m0 + 16, n0 + 16, bn1, out, ldo, hi, l15);
  store_tile_relu_bf16(c12, m0 + 16, n0 + 32, bn2, out, ldo, hi, l15);
  store_tile_relu_bf16(c13, m0 + 16, n0 + 48, bn3, out, ldo, hi, l15);
}

// ---------- thin GEMM (N=16), up to two accumulation phases ----------
// mode 0: sigmoid -> bf16 scores ; mode 1: softmax over 16 experts -> f32 probs
__global__ __launch_bounds__(256) void gemm_n16_kernel(const u16* __restrict__ A1, int lda1,
                                                       const u16* __restrict__ BT1, int ldb1, int K1,
                                                       const u16* __restrict__ A2, int lda2,
                                                       const u16* __restrict__ BT2, int ldb2, int K2,
                                                       const float* __restrict__ bias, int mode,
                                                       u16* __restrict__ out_bf16,
                                                       float* __restrict__ out_f32) {
  const int lane = threadIdx.x & 31;
  const int wid  = threadIdx.x >> 5;
  const int l15 = lane & 15, hi = lane >> 4;
  const int m0 = blockIdx.x * 128 + wid * 16;

  v8f c = {};
  mma_loop(A1 + (size_t)(m0 + l15) * lda1 + hi * 8,
           BT1 + (size_t)l15 * ldb1 + hi * 8, K1, c);
  if (A2)
    mma_loop(A2 + (size_t)(m0 + l15) * lda2 + hi * 8,
             BT2 + (size_t)l15 * ldb2 + hi * 8, K2, c);

  float bn = bias[l15];
  Acc u; u.v = c;
  if (mode == 0) {
#pragma unroll
    for (int r = 0; r < 8; ++r) {
      int m = m0 + r + hi * 8;
      float s = 1.f / (1.f + __expf(-(u.f[r] + bn)));
      out_bf16[(size_t)m * NE + l15] = f2bf(s);
    }
  } else {
#pragma unroll
    for (int r = 0; r < 8; ++r) {
      float v = u.f[r] + bn;
      float mx = v;
      for (int o = 1; o < 16; o <<= 1) mx = fmaxf(mx, __shfl_xor(mx, o));
      float e = __expf(v - mx);
      float s = e;
      for (int o = 1; o < 16; o <<= 1) s += __shfl_xor(s, o);
      int m = m0 + r + hi * 8;
      out_f32[(size_t)m * NE + l15] = e / s;
    }
  }
}

// ---------- top-4 + masked-softmax gates ----------
__global__ __launch_bounds__(256) void topk_kernel(const float* __restrict__ probs,
                                                   const int* __restrict__ kptr,
                                                   float* __restrict__ gates,
                                                   float* __restrict__ idxout) {
  int r = blockIdx.x * 256 + threadIdx.x;
  if (r >= NB) return;
  float p[16];
#pragma unroll
  for (int i = 0; i < 16; ++i) p[i] = probs[(size_t)r * NE + i];
  int k = *kptr;
  float tp[4]; int ti[4];
#pragma unroll
  for (int j = 0; j < 4; ++j) {
    float best = -1e30f; int bi = 0;
#pragma unroll
    for (int i = 0; i < 16; ++i)
      if (p[i] > best) { best = p[i]; bi = i; }
    tp[j] = best; ti[j] = bi; p[bi] = -1e31f;
  }
  float mx = tp[0];
  float e[4], s = 0.f;
#pragma unroll
  for (int j = 0; j < 4; ++j) { e[j] = (j < k) ? __expf(tp[j] - mx) : 0.f; s += e[j]; }
#pragma unroll
  for (int j = 0; j < 4; ++j) {
    gates[(size_t)r * 4 + j]  = e[j] / s;
    idxout[(size_t)r * 4 + j] = (float)ti[j];
  }
}

// ---------- launch ----------
extern "C" void kernel_launch(void* const* d_in, const int* in_sizes, int n_in,
                              void* d_out, int out_size, void* d_ws, size_t ws_size,
                              hipStream_t stream) {
  (void)in_sizes; (void)n_in; (void)out_size; (void)ws_size;

  const float* x   = (const float*)d_in[0];
  const float* pat = (const float*)d_in[1];
  const float* Ws1 = (const float*)d_in[2];
  const float* bs1 = (const float*)d_in[3];
  const float* Ws2 = (const float*)d_in[4];
  const float* bs2 = (const float*)d_in[5];
  const float* Wr1 = (const float*)d_in[6];
  const float* br1 = (const float*)d_in[7];
  const float* Wr2 = (const float*)d_in[8];
  const float* br2 = (const float*)d_in[9];
  const float* Wk1 = (const float*)d_in[10];
  const float* bk1 = (const float*)d_in[11];
  const float* Wk2 = (const float*)d_in[12];
  const float* bk2 = (const float*)d_in[13];

  char* ws = (char*)d_ws;
  u16*  xb    = (u16*)(ws + OFF_XB);
  u16*  Ws1T  = (u16*)(ws + OFF_WS1T);
  u16*  Wr1T  = (u16*)(ws + OFF_WR1T);
  u16*  Ws2T  = (u16*)(ws + OFF_WS2T);
  u16*  Wr2T  = (u16*)(ws + OFF_WR2T);
  u16*  patB  = (u16*)(ws + OFF_PATB);
  u16*  scr   = (u16*)(ws + OFF_SCR);
  u16*  hbuf  = (u16*)(ws + OFF_H);
  int*  kval  = (int*)(ws + OFF_KVAL);

  float* out   = (float*)d_out;
  float* gates = out + OOF_GATES;
  float* idxo  = out + OOF_IDX;
  float* probs = out + OOF_PROBS;
  float* stats = out + OOF_STATS;

  // bf16 conversion / weight transposition
  cvt_kernel <<<4096, 256, 0, stream>>>(x,   xb,   NB * ND);
  cvt_kernel <<<64,   256, 0, stream>>>(pat, patB, NE * ND);
  cvtT_kernel<<<2048, 256, 0, stream>>>(Ws1, Ws1T, ND,  NH);
  cvtT_kernel<<<2048, 256, 0, stream>>>(Wr1, Wr1T, NDE, NH);
  cvtT_kernel<<<64,   256, 0, stream>>>(Ws2, Ws2T, NH,  NE);
  cvtT_kernel<<<64,   256, 0, stream>>>(Wr2, Wr2T, NH,  NE);

  // statistics + dynamic-k (median floor via bucket counts)
  stats_kernel<<<NB, 256, 0, stream>>>(x, stats);
  kpred_kernel<<<1, 256, 0, stream>>>(stats, Wk1, bk1, Wk2, bk2, kval);

  dim3 gBig(NB / 128, NH / 128);
  // h1 = relu(x @ Ws1 + bs1)
  gemm_big_kernel<<<gBig, 256, 0, stream>>>(xb, ND, Ws1T, ND, bs1, hbuf, NH, ND,
                                            nullptr, nullptr);
  // scores = sigmoid(x @ pat^T + h1 @ Ws2 + bs2)
  gemm_n16_kernel<<<NB / 128, 256, 0, stream>>>(xb, ND, patB, ND, ND,
                                                hbuf, NH, Ws2T, NH, NH,
                                                bs2, 0, scr, nullptr);
  // h2 = relu(x @ Wr1[:D] + scores @ Wr1[D:] + br1)  (overwrites h1)
  gemm_big_kernel<<<gBig, 256, 0, stream>>>(xb, ND, Wr1T, NDE, br1, hbuf, NH, ND,
                                            scr, Wr1T + ND);
  // probs = softmax(h2 @ Wr2 + br2)
  gemm_n16_kernel<<<NB / 128, 256, 0, stream>>>(hbuf, NH, Wr2T, NH, NH,
                                                nullptr, 0, nullptr, 0, 0,
                                                br2, 1, nullptr, probs);
  // gates + indices
  topk_kernel<<<NB / 256, 256, 0, stream>>>(probs, kval, gates, idxo);
}
